// OneHiddenLayer_81106162418172
// MI455X (gfx1250) — matrix operand order repaired and checked by reading, not compiled, not run on hardware
//
#include <hip/hip_runtime.h>
#include <stdint.h>

typedef __bf16 bf16;
typedef __attribute__((ext_vector_type(16))) __bf16 v16bf;
typedef __attribute__((ext_vector_type(8)))  __bf16 v8bf;
typedef __attribute__((ext_vector_type(8)))  float  v8f;

#define DIM       4096
#define PLANE     ((size_t)DIM * DIM)   // elements per plane
#define TILE_M    128
#define TILE_N    128
#define TILE_K    64                    // 2 WMMA K-slabs per stage
#define PITCH     72                    // bf16 per LDS row (64 + 8 pad), 144B: 16B aligned
#define EPSV      0.01f
#define NTHREADS  128                   // 4 waves, each owns a 64x64 output tile

// ---- CDNA5 async global -> LDS copy (ASYNCcnt path), 16B per lane ----------
__device__ __forceinline__ void asyncCopy16(void* ldsPtr, const void* gPtr) {
    uint32_t ldsOff = (uint32_t)(uintptr_t)ldsPtr;   // low 32 bits = LDS byte addr
    uint64_t gAddr  = (uint64_t)(uintptr_t)gPtr;
    asm volatile("global_load_async_to_lds_b128 %0, %1, off"
                 :: "v"(ldsOff), "v"(gAddr) : "memory");
}
__device__ __forceinline__ void waitAsync0() {
    asm volatile("s_wait_asynccnt 0x0" ::: "memory");
}

// ---- assemble a 16-element bf16 fragment from two 16B LDS chunks -----------
union FragU { v16bf v; v8bf h[2]; };
__device__ __forceinline__ v16bf ldsFrag(const bf16* p0, const bf16* p1) {
    FragU f;
    f.h[0] = *(const v8bf*)p0;
    f.h[1] = *(const v8bf*)p1;
    return f.v;
}

// ---- O(N^2) preprocessing: fp32 plane -> bf16 hi/lo planes -----------------
__global__ __launch_bounds__(256)
void split_fp32(const float* __restrict__ in, bf16* __restrict__ hi,
                bf16* __restrict__ lo) {
    size_t idx = ((size_t)blockIdx.x * 256 + threadIdx.x) * 8;
    float4 a = *(const float4*)(in + idx);
    float4 b = *(const float4*)(in + idx + 4);
    float f[8] = {a.x, a.y, a.z, a.w, b.x, b.y, b.z, b.w};
    v8bf vh, vl;
#pragma unroll
    for (int i = 0; i < 8; ++i) {
        bf16 h = (bf16)f[i];               // RNE
        vh[i] = h;
        vl[i] = (bf16)(f[i] - (float)h);   // residual
    }
    *(v8bf*)(hi + idx) = vh;
    *(v8bf*)(lo + idx) = vl;
}

// ---- C = act( A @ B^T + bias ), operands pre-split as bf16 hi/lo planes ----
// MODE 0: write C as bf16 hi/lo planes (feeds next GEMM)
// MODE 1: C = flipcos(acc + bias) as fp32
template <int MODE>
__global__ __launch_bounds__(NTHREADS)
void gemm_bf16x3(const bf16* __restrict__ Ahi, const bf16* __restrict__ Alo,
                 const bf16* __restrict__ Bhi, const bf16* __restrict__ Blo,
                 const float* __restrict__ bias,
                 float* __restrict__ Cf, bf16* __restrict__ Chi,
                 bf16* __restrict__ Clo) {
    // planes: 0=Ahi 1=Alo 2=Bhi 3=Blo  (tile 128 x 64 bf16 each, double buffered)
    __shared__ bf16 sh[2][4][TILE_M * PITCH];   // 144 KB

    const int tid   = threadIdx.x;
    const int lane  = tid & 31;
    const int wid   = tid >> 5;        // 4 waves
    const int waveM = wid & 1;         // 2 bands of 64 rows
    const int waveN = wid >> 1;        // 2 bands of 64 cols
    const int half  = lane >> 4;
    const int l16   = lane & 15;

    const int mBase = blockIdx.y * TILE_M;
    const int nBase = blockIdx.x * TILE_N;

    // staging: 64-bf16 row = 128B = 8 chunks of 16B; 128 threads cover 16 rows
    const int cRow  = tid & 7;          // chunk in row
    const int rBase = tid >> 3;         // 0..15; rows rBase + 16*j, j=0..7

    const size_t gOffA = (size_t)(mBase + rBase) * DIM + cRow * 8;
    const size_t gOffB = (size_t)(nBase + rBase) * DIM + cRow * 8;

    auto stage = [&](int buf, int kt) {
        const size_t ko = (size_t)kt * TILE_K;
        const bf16* s0 = Ahi + gOffA + ko;
        const bf16* s1 = Alo + gOffA + ko;
        const bf16* s2 = Bhi + gOffB + ko;
        const bf16* s3 = Blo + gOffB + ko;
        bf16* d = &sh[buf][0][rBase * PITCH + cRow * 8];
        const size_t planeStride = (size_t)TILE_M * PITCH;
#pragma unroll
        for (int j = 0; j < 8; ++j) {
            const size_t go = (size_t)j * 16 * DIM;
            bf16* dj = d + j * 16 * PITCH;
            asyncCopy16(dj + 0 * planeStride, s0 + go);
            asyncCopy16(dj + 1 * planeStride, s1 + go);
            asyncCopy16(dj + 2 * planeStride, s2 + go);
            asyncCopy16(dj + 3 * planeStride, s3 + go);
        }
    };

    v8f acc[4][4];
#pragma unroll
    for (int i = 0; i < 4; ++i)
#pragma unroll
        for (int j = 0; j < 4; ++j)
            acc[i][j] = (v8f){0.f, 0.f, 0.f, 0.f, 0.f, 0.f, 0.f, 0.f};

    stage(0, 0);

    const int KT = DIM / TILE_K;
    for (int kt = 0; kt < KT; ++kt) {
        const int buf = kt & 1;
        waitAsync0();
        __syncthreads();
        if (kt + 1 < KT) stage(buf ^ 1, kt + 1);

#pragma unroll
        for (int s = 0; s < 2; ++s) {              // two K=32 slabs
            // A fragments: K(e) = 16*(e/8) + 8*half + e%8  (within slab)
            v16bf ahi[4], alo[4];
#pragma unroll
            for (int fm = 0; fm < 4; ++fm) {
                const int row = waveM * 64 + fm * 16 + l16;
                const bf16* rh = &sh[buf][0][row * PITCH + s * 32];
                const bf16* rl = &sh[buf][1][row * PITCH + s * 32];
                ahi[fm] = ldsFrag(rh + 8 * half, rh + 16 + 8 * half);
                alo[fm] = ldsFrag(rl + 8 * half, rl + 16 + 8 * half);
            }
#pragma unroll
            for (int fn = 0; fn < 4; ++fn) {
                // B fragment: lane col = l16, K(e) = 16*(e/8) + 8*half + e%8 (within slab)
                const int row = waveN * 64 + fn * 16 + l16;
                const bf16* rh = &sh[buf][2][row * PITCH + s * 32];
                const bf16* rl = &sh[buf][3][row * PITCH + s * 32];
                v16bf bhi = ldsFrag(rh + 8 * half, rh + 16 + 8 * half);
                v16bf blo = ldsFrag(rl + 8 * half, rl + 16 + 8 * half);
#pragma unroll
                for (int fm = 0; fm < 4; ++fm) {
                    acc[fm][fn] = __builtin_amdgcn_wmma_f32_16x16x32_bf16(
                        false, ahi[fm], false, bhi, (short)0, acc[fm][fn], false, false);
                    acc[fm][fn] = __builtin_amdgcn_wmma_f32_16x16x32_bf16(
                        false, ahi[fm], false, blo, (short)0, acc[fm][fn], false, false);
                    acc[fm][fn] = __builtin_amdgcn_wmma_f32_16x16x32_bf16(
                        false, alo[fm], false, bhi, (short)0, acc[fm][fn], false, false);
                }
            }
        }
    }

    // ---- epilogue: C/D layout: lane col = l16, rows 8*half + r -------------
    const int colBase = nBase + waveN * 64;
#pragma unroll
    for (int fn = 0; fn < 4; ++fn) {
        const int col = colBase + fn * 16 + l16;
        const float bv = bias[col];
#pragma unroll
        for (int fm = 0; fm < 4; ++fm) {
            const size_t row0 = mBase + waveM * 64 + fm * 16 + half * 8;
#pragma unroll
            for (int r = 0; r < 8; ++r) {
                float v = acc[fm][fn][r] + bv;
                const size_t o = (row0 + r) * DIM + col;
                if (MODE == 1) {
                    v = cosf(v);
                    v = (fabsf(v) < EPSV) ? -v : v;
                    Cf[o] = v;
                } else {
                    bf16 h = (bf16)v;
                    Chi[o] = h;
                    Clo[o] = (bf16)(v - (float)h);
                }
            }
        }
    }
}

extern "C" void kernel_launch(void* const* d_in, const int* in_sizes, int n_in,
                              void* d_out, int out_size, void* d_ws, size_t ws_size,
                              hipStream_t stream) {
    const float* x      = (const float*)d_in[0];
    const float* W      = (const float*)d_in[1];
    const float* b      = (const float*)d_in[2];
    const float* g      = (const float*)d_in[3];
    const float* g_bias = (const float*)d_in[4];
    float* out = (float*)d_out;

    // workspace: 6 bf16 planes (192 MB)
    bf16* p = (bf16*)d_ws;
    bf16* xh = p + 0 * PLANE;  bf16* xl = p + 1 * PLANE;   // later reused for g
    bf16* wh = p + 2 * PLANE;  bf16* wl = p + 3 * PLANE;
    bf16* hh = p + 4 * PLANE;  bf16* hl = p + 5 * PLANE;

    const int splitBlocks = (int)(PLANE / (256 * 8));      // 8192
    dim3 grid(DIM / TILE_N, DIM / TILE_M);

    split_fp32<<<splitBlocks, 256, 0, stream>>>(x, xh, xl);
    split_fp32<<<splitBlocks, 256, 0, stream>>>(W, wh, wl);
    gemm_bf16x3<0><<<grid, NTHREADS, 0, stream>>>(xh, xl, wh, wl, b,
                                                  nullptr, hh, hl);
    split_fp32<<<splitBlocks, 256, 0, stream>>>(g, xh, xl); // x planes now free
    gemm_bf16x3<1><<<grid, NTHREADS, 0, stream>>>(hh, hl, xh, xl, g_bias,
                                                  out, nullptr, nullptr);
}
